// TripletCorrelationalDetector_67448166416866
// MI455X (gfx1250) — matrix operand order repaired and checked by reading, not compiled, not run on hardware
//
#include <hip/hip_runtime.h>
#include <hip/hip_bf16.h>
#include <stdint.h>

// ---------------------------------------------------------------------------
// Types for CDNA5 WMMA / TDM
// ---------------------------------------------------------------------------
typedef __attribute__((ext_vector_type(16))) __bf16 v16bf;
typedef __attribute__((ext_vector_type(8)))  float  v8f;
typedef __attribute__((ext_vector_type(4)))  unsigned int u32x4;
typedef __attribute__((ext_vector_type(8)))  unsigned int u32x8;

static __device__ __forceinline__ v8f wmma_bf16(v16bf a, v16bf b, v8f c) {
  // D = A(16x32 bf16) * B(32x16 bf16) + C(16x16 f32)
  return __builtin_amdgcn_wmma_f32_16x16x32_bf16(false, a, false, b,
                                                 (short)0, c, false, false);
}

static __device__ __forceinline__ unsigned short f2bf(float f) {
  union { float f; uint32_t u; } c; c.f = f;
  uint32_t u = c.u;
  uint32_t r = u + 0x7FFFu + ((u >> 16) & 1u);   // round-to-nearest-even
  return (unsigned short)(r >> 16);
}

// Async global->LDS copy of one 16-byte chunk (gfx1250, tracked by ASYNCcnt).
// VDST holds the LDS byte address (= low 32 bits of generic ptr);
// VADDR is the 64-bit global address.
static __device__ __forceinline__ void async_g2l_b128(const void* gptr, void* lptr) {
  unsigned lds = (unsigned)(uintptr_t)lptr;
  unsigned long long ga = (unsigned long long)(uintptr_t)gptr;
  asm volatile("global_load_async_to_lds_b128 %0, %1, off"
               :: "v"(lds), "v"(ga) : "memory");
}
static __device__ __forceinline__ void wait_async0() {
  asm volatile("s_wait_asynccnt 0x0" ::: "memory");
}

// ---------------------------------------------------------------------------
// Kernel 1: broadcast anchor [3,127,127] -> d_out[64,3,127,127]
// ---------------------------------------------------------------------------
__global__ void bcast_anchor(const float* __restrict__ a, float* __restrict__ out,
                             int per, long long total) {
  long long i = (long long)blockIdx.x * blockDim.x + threadIdx.x;
  long long stride = (long long)gridDim.x * blockDim.x;
  for (; i < total; i += stride) out[i] = a[i % per];
}

// ---------------------------------------------------------------------------
// Kernel 2: im2col directly into WMMA A-fragment order (bf16).
// col layout: [mtile][kblk][lane(32)][j(16)] ushort, 1KB per 16x32 tile.
// A 16-bit 16x32 layout (ISA 7.12.2): lane<16 -> M=lane, K = j<8 ? j : j+8
//                                     lane>=16 -> M=lane-16, K = 8 + (j<8 ? j : j+8)
// ---------------------------------------------------------------------------
template<int CIN, int S, bool SRC_NCHW>
__global__ void im2col_frag(const float* __restrict__ src, unsigned short* __restrict__ col,
                            int B, int IH, int IW, int OH, int OW,
                            int KB, int M, int MT) {
  int t    = blockIdx.x * blockDim.x + threadIdx.x;
  int lane = t & 31;
  int kb   = (t >> 5) % KB;
  int mt   = (t >> 5) / KB;
  if (mt >= MT) return;

  int m = mt * 16 + (lane & 15);
  bool mvalid = (m < M);
  int n = 0, oy = 0, ox = 0;
  if (mvalid) {
    n = m / (OH * OW);
    int r = m - n * OH * OW;
    oy = r / OW; ox = r - oy * OW;
  }
  __align__(16) unsigned short tmp[16];
#pragma unroll
  for (int j = 0; j < 16; ++j) {
    int kl = ((lane >= 16) ? 8 : 0) + ((j < 8) ? j : j + 8);
    int K  = kb * 32 + kl;
    float v = 0.0f;
    if (mvalid && K < 9 * CIN) {
      int patch = K / CIN;
      int ic    = K - patch * CIN;
      int ky = patch / 3, kx = patch - ky * 3;
      int iy = oy * S + ky, ix = ox * S + kx;
      size_t idx = SRC_NCHW
          ? ((size_t)((n * CIN + ic) * IH + iy) * IW + ix)
          : ((size_t)((n * IH + iy) * IW + ix) * CIN + ic);
      v = src[idx];
    }
    tmp[j] = f2bf(v);
  }
  uint4* dst = reinterpret_cast<uint4*>(col + ((size_t)(mt * KB + kb) * 32 + lane) * 16);
  dst[0] = *reinterpret_cast<uint4*>(tmp);
  dst[1] = *reinterpret_cast<uint4*>(tmp + 8);
}

// ---------------------------------------------------------------------------
// Kernel 3: pack conv weights [O,I,3,3] f32 -> WMMA B-fragment order (bf16).
// B 16-bit 32x16 layout: lane<16 -> N=lane,   K = j<8 ? j : j+8
//                        lane>=16 -> N=lane-16, K = 8 + (j<8 ? j : j+8)
// wfrag layout: [octile][kblk][lane(32)][j(16)]
// ---------------------------------------------------------------------------
__global__ void pack_weights(const float* __restrict__ w, unsigned short* __restrict__ wf,
                             int CIN, int COUT, int NPAD, int KPAD) {
  int t    = blockIdx.x * blockDim.x + threadIdx.x;
  int lane = t & 31;
  int KBn  = KPAD / 32;
  int kb   = (t >> 5) % KBn;
  int ot   = (t >> 5) / KBn;
  if (ot >= NPAD / 16) return;

  int oc = ot * 16 + (lane & 15);
  __align__(16) unsigned short tmp[16];
#pragma unroll
  for (int j = 0; j < 16; ++j) {
    int K = kb * 32 + ((lane >= 16) ? 8 : 0) + ((j < 8) ? j : j + 8);
    float v = 0.0f;
    if (oc < COUT && K < 9 * CIN) {
      int patch = K / CIN;
      int ic    = K - patch * CIN;
      int ky = patch / 3, kx = patch - ky * 3;
      v = w[((size_t)(oc * CIN + ic) * 3 + ky) * 3 + kx];
    }
    tmp[j] = f2bf(v);
  }
  uint4* dst = reinterpret_cast<uint4*>(wf + ((size_t)(ot * KBn + kb) * 32 + lane) * 16);
  dst[0] = *reinterpret_cast<uint4*>(tmp);
  dst[1] = *reinterpret_cast<uint4*>(tmp + 8);
}

// ---------------------------------------------------------------------------
// Kernel 4: WMMA GEMM. One block (4 waves) -> 128 pixels x (NT*16) channels.
// Each wave owns TWO 16-pixel M-tiles so every B fragment staged in LDS feeds
// two WMMAs (halves ds traffic per WMMA). B slices are staged per-block via
// async global->LDS (double buffered, ASYNCcnt). A is streamed with b128.
// Stores: unguarded (convOut has Mpad rows), COUT compile-time so r*COUT
// folds into the store immediate offset.
// ---------------------------------------------------------------------------
template<int NT, int KB, int COUT>
__global__ void __launch_bounds__(128)
conv_gemm(const unsigned short* __restrict__ col,
          const unsigned short* __restrict__ wf,
          float* __restrict__ out) {
  __shared__ __align__(16) unsigned short Bsh[2][NT * 512];

  int lane = threadIdx.x & 31;
  int wave = threadIdx.x >> 5;
  int mt0  = (blockIdx.x * 4 + wave) * 2;

  // cooperative async stage of the B slice for one k-block: NT KB of 1KB each
  auto stage_B = [&](int kb, int buf) {
#pragma unroll
    for (int i = threadIdx.x; i < NT * 64; i += 128) {   // 16B chunks
      int nt = i >> 6, w16 = i & 63;
      async_g2l_b128(wf + ((size_t)(nt * KB + kb) * 512 + w16 * 8),
                     &Bsh[buf][nt * 512 + w16 * 8]);
    }
  };

  v8f acc0[NT] = {};
  v8f acc1[NT] = {};

  stage_B(0, 0);
  for (int kb = 0; kb < KB; ++kb) {
    int buf = kb & 1;
    // two A fragments (consecutive m-tiles), streamed; read exactly once
    union { v16bf v; uint4 q[2]; } A0, A1;
    const uint4* ap = reinterpret_cast<const uint4*>(col) +
                      ((size_t)(mt0 * KB + kb) * 32 + lane) * 2;
    A0.q[0] = ap[0];
    A0.q[1] = ap[1];
    const uint4* ap1 = ap + (size_t)KB * 64;   // next m-tile, same kb
    A1.q[0] = ap1[0];
    A1.q[1] = ap1[1];
    if (kb + 1 < KB) __builtin_prefetch(ap + 64, 0, 0);  // next A tile

    wait_async0();       // B slice for kb resident in LDS
    __syncthreads();
    if (kb + 1 < KB) stage_B(kb + 1, buf ^ 1);

#pragma unroll
    for (int nt = 0; nt < NT; ++nt) {
      union { v16bf v; uint4 q[2]; } Bf;
      const uint4* bp = reinterpret_cast<const uint4*>(&Bsh[buf][nt * 512 + lane * 16]);
      Bf.q[0] = bp[0];
      Bf.q[1] = bp[1];
      acc0[nt] = wmma_bf16(A0.v, Bf.v, acc0[nt]);
      acc1[nt] = wmma_bf16(A1.v, Bf.v, acc1[nt]);
    }
    __syncthreads();     // all reads of buf done before it is overwritten
  }

  // C/D layout: VGPR r -> M=r (lanes 0-15) or M=8+r (lanes 16-31); N = lane&15
  int mbase = mt0 * 16 + ((lane >= 16) ? 8 : 0);
  int ncol  = lane & 15;
#pragma unroll
  for (int nt = 0; nt < NT; ++nt) {
    int oc = nt * 16 + ncol;
    if (NT * 16 == COUT || oc < COUT) {
      float* op = out + (size_t)mbase * COUT + oc;
#pragma unroll
      for (int r = 0; r < 8; ++r) op[(size_t)r * COUT] = acc0[nt][r];
      float* op1 = op + (size_t)16 * COUT;
#pragma unroll
      for (int r = 0; r < 8; ++r) op1[(size_t)r * COUT] = acc1[nt][r];
    }
  }
}

// ---------------------------------------------------------------------------
// BatchNorm (training mode): sum/sumsq -> scale/shift -> in-place apply
// ---------------------------------------------------------------------------
__global__ void zerof(float* p, int n) {
  int i = blockIdx.x * blockDim.x + threadIdx.x;
  if (i < n) p[i] = 0.0f;
}

__global__ void bn_stats(const float* __restrict__ x, float* __restrict__ sum,
                         float* __restrict__ sumsq, int M, int C) {
  int t = blockIdx.x * blockDim.x + threadIdx.x;
  int total = gridDim.x * blockDim.x;      // launched as C blocks * 256 => total % C == 0
  int c = t % C;
  int r0 = t / C;
  int rstep = total / C;
  float s = 0.0f, q = 0.0f;
  for (int r = r0; r < M; r += rstep) {
    float v = x[(size_t)r * C + c];
    s += v; q += v * v;
  }
  atomicAdd(&sum[c], s);
  atomicAdd(&sumsq[c], q);
}

__global__ void bn_finalize(const float* __restrict__ sum, const float* __restrict__ sumsq,
                            const float* __restrict__ g, const float* __restrict__ b,
                            float* __restrict__ scale, float* __restrict__ shift,
                            int C, float invCnt) {
  int c = blockIdx.x * blockDim.x + threadIdx.x;
  if (c < C) {
    float m  = sum[c] * invCnt;
    float v  = sumsq[c] * invCnt - m * m;
    float rs = rsqrtf(v + 1e-5f);
    float sc = g[c] * rs;
    scale[c] = sc;
    shift[c] = b[c] - m * sc;
  }
}

__global__ void bn_apply(float* __restrict__ x, const float* __restrict__ scale,
                         const float* __restrict__ shift, long long n, int C, int relu) {
  long long i = (long long)blockIdx.x * blockDim.x + threadIdx.x;
  long long stride = (long long)gridDim.x * blockDim.x;
  for (; i < n; i += stride) {
    int c = (int)(i % C);
    float v = x[i] * scale[c] + shift[c];
    if (relu) v = fmaxf(v, 0.0f);
    x[i] = v;
  }
}

// ---------------------------------------------------------------------------
// Cross-correlation: f[B,28,28,128] (NHWC) with k[12,12,128] -> out[B,17,17]
// One block per output pixel. The 73.7KB anchor-feature tensor is staged into
// LDS once per block by the Tensor Data Mover (tensor_load_to_lds, D# in
// SGPRs, TENSORcnt-synchronized), then K=18432 float4 dot + LDS reduce.
// ---------------------------------------------------------------------------
__global__ void __launch_bounds__(256)
xcorr_kernel(const float* __restrict__ f, const float* __restrict__ k,
             float* __restrict__ out) {
  __shared__ __align__(16) float ksh[12 * 12 * 128];   // 73728 B
  __shared__ float red[256];

  if (threadIdx.x < 32) {          // wave 0 issues one TDM op for the block
    unsigned lds = (unsigned)(uintptr_t)ksh;
    unsigned long long ga = (unsigned long long)(uintptr_t)k;
    const unsigned D = 12 * 12 * 128;                    // 18432 f32 elements
    union { u32x4 v; unsigned u[4]; } g0 = {};
    g0.u[0] = 1u;                                        // count=1 (valid user D#)
    g0.u[1] = lds;                                       // lds_addr
    g0.u[2] = (unsigned)ga;                              // global_addr[31:0]
    g0.u[3] = (unsigned)((ga >> 32) & 0x01FFFFFFu) | 0x80000000u; // addr[56:32]|type=2
    union { u32x8 v; unsigned u[8]; } g1 = {};
    g1.u[0] = 2u << 16;                                  // data_size = 4 bytes
    g1.u[1] = (D & 0xFFFFu) << 16;                       // tensor_dim0[15:0]
    g1.u[2] = (D >> 16) | (1u << 16);                    // tensor_dim0[31:16], dim1=1
    g1.u[3] = (D & 0xFFFFu) << 16;                       // tile_dim0 = 18432 (1-D tile)
    g1.u[4] = 0u;                                        // tile_dim1/2 unused
    g1.u[5] = D;                                         // tensor_dim0_stride[31:0]
    g1.u[6] = 0u;
    g1.u[7] = 0u;
    asm volatile("tensor_load_to_lds %0, %1" :: "s"(g0.v), "s"(g1.v) : "memory");
    __builtin_amdgcn_s_wait_tensorcnt(0);
  }
  __syncthreads();

  int blk = blockIdx.x;
  int b = blk / 289;
  int r = blk - b * 289;
  int oy = r / 17, ox = r - oy * 17;

  float acc = 0.0f;
  for (int ky = 0; ky < 12; ++ky) {
    const float4* fr = reinterpret_cast<const float4*>(
        f + (((size_t)b * 28 + oy + ky) * 28 + ox) * 128);
    const float4* kr = reinterpret_cast<const float4*>(ksh + ky * 12 * 128);
    for (int i = threadIdx.x; i < 384; i += 256) {
      float4 a = fr[i], w = kr[i];
      acc += a.x * w.x + a.y * w.y + a.z * w.z + a.w * w.w;
    }
  }
  red[threadIdx.x] = acc;
  __syncthreads();
  for (int s = 128; s > 0; s >>= 1) {
    if (threadIdx.x < s) red[threadIdx.x] += red[threadIdx.x + s];
    __syncthreads();
  }
  if (threadIdx.x == 0) out[blk] = red[0];
}

// ---------------------------------------------------------------------------
// Host-side orchestration
// ---------------------------------------------------------------------------
struct LayerCfg { int CIN, COUT, NPAD, KPAD, S, NT, KB; };
static const LayerCfg LC[5] = {
  { 3,   3,  16,  32, 2, 1,  1},
  { 3,  16,  16,  32, 2, 1,  1},
  {16,  32,  32, 160, 1, 2,  5},
  {32,  64,  64, 288, 2, 4,  9},
  {64, 128, 128, 576, 1, 8, 18},
};
static const size_t WOFF[5] = {0, 512, 1024, 6144, 24576};  // ushort elems

struct Scratch {
  unsigned short* col;
  float*          convOut;
  unsigned short* wfrag;
  float*          sum;    // [128]
  float*          sumsq;  // [128]
  float*          scale;  // [128]
  float*          shift;  // [128]
  float*          ancF;   // [12*12*128]
};

static void run_branch(hipStream_t s, const float* input_nchw, int B, int IH0, int IW0,
                       const Scratch& P, void* const* d_in) {
  int ih = IH0, iw = IW0;
  for (int l = 0; l < 5; ++l) {
    const LayerCfg& c = LC[l];
    int oh = (ih - 3) / c.S + 1;
    int ow = (iw - 3) / c.S + 1;
    int M    = B * oh * ow;
    int Mpad = (M + 127) & ~127;     // 128 pixels per block (2 m-tiles per wave)
    int MT   = Mpad / 16;
    int KB   = c.KPAD / 32;

    // --- im2col into A-fragment order (layer 0 reads NCHW input, rest NHWC f32)
    int ithreads = MT * KB * 32;
    dim3 ig((ithreads + 255) / 256);
    switch (l) {
      case 0: im2col_frag< 3, 2, true ><<<ig, 256, 0, s>>>(input_nchw, P.col, B, ih, iw, oh, ow, KB, M, MT); break;
      case 1: im2col_frag< 3, 2, false><<<ig, 256, 0, s>>>(P.convOut,  P.col, B, ih, iw, oh, ow, KB, M, MT); break;
      case 2: im2col_frag<16, 1, false><<<ig, 256, 0, s>>>(P.convOut,  P.col, B, ih, iw, oh, ow, KB, M, MT); break;
      case 3: im2col_frag<32, 2, false><<<ig, 256, 0, s>>>(P.convOut,  P.col, B, ih, iw, oh, ow, KB, M, MT); break;
      case 4: im2col_frag<64, 1, false><<<ig, 256, 0, s>>>(P.convOut,  P.col, B, ih, iw, oh, ow, KB, M, MT); break;
    }

    // --- WMMA GEMM: col x wfrag -> convOut[Mpad rows] (prev features were
    //     fully consumed by the im2col above; pad rows are never read)
    const unsigned short* wf = P.wfrag + WOFF[l];
    dim3 gg(Mpad / 128);
    switch (l) {
      case 0: conv_gemm<1,  1,   3><<<gg, 128, 0, s>>>(P.col, wf, P.convOut); break;
      case 1: conv_gemm<1,  1,  16><<<gg, 128, 0, s>>>(P.col, wf, P.convOut); break;
      case 2: conv_gemm<2,  5,  32><<<gg, 128, 0, s>>>(P.col, wf, P.convOut); break;
      case 3: conv_gemm<4,  9,  64><<<gg, 128, 0, s>>>(P.col, wf, P.convOut); break;
      case 4: conv_gemm<8, 18, 128><<<gg, 128, 0, s>>>(P.col, wf, P.convOut); break;
    }

    // --- BatchNorm (batch statistics). Conv bias cancels exactly under BN.
    zerof<<<1, 256, 0, s>>>(P.sum, 256);  // sum[128] + sumsq[128] contiguous
    bn_stats<<<c.COUT, 256, 0, s>>>(P.convOut, P.sum, P.sumsq, M, c.COUT);
    const float* g = (const float*)d_in[13 + l];
    const float* b = (const float*)d_in[18 + l];
    bn_finalize<<<1, 128, 0, s>>>(P.sum, P.sumsq, g, b, P.scale, P.shift,
                                  c.COUT, 1.0f / (float)M);
    long long n = (long long)M * c.COUT;
    int nblk = (int)(((n + 255) / 256) < 4096 ? ((n + 255) / 256) : 4096);
    bn_apply<<<nblk, 256, 0, s>>>(P.convOut, P.scale, P.shift, n, c.COUT, l < 4 ? 1 : 0);

    ih = oh; iw = ow;
  }
}

extern "C" void kernel_launch(void* const* d_in, const int* in_sizes, int n_in,
                              void* d_out, int out_size, void* d_ws, size_t ws_size,
                              hipStream_t stream) {
  (void)in_sizes; (void)n_in; (void)out_size; (void)ws_size;

  const float* pos    = (const float*)d_in[0];
  const float* neg    = (const float*)d_in[1];
  const float* anchor = (const float*)d_in[2];
  float* out = (float*)d_out;

  // Workspace layout (~113 MiB):
  uint8_t* ws = (uint8_t*)d_ws;
  Scratch P;
  P.col     = (unsigned short*)(ws);                 // 80 MiB (max: L3 pos 76.2 MB)
  P.convOut = (float*)(ws + 0x5000000);              // 32 MiB (max: L3 pos 30.5 MB)
  P.wfrag   = (unsigned short*)(ws + 0x7000000);     // 256 KiB (98304 elems used)
  float* st = (float*)(ws + 0x7040000);
  P.sum   = st;       P.sumsq = st + 128;
  P.scale = st + 256; P.shift = st + 384;
  P.ancF  = (float*)(ws + 0x7041000);                // 12*12*128 f32

  // ---- output offsets: (anchor_b, cxp, cxn)
  const int ANCH_PER = 3 * 127 * 127;                // 48387
  const long long ANCH_TOTAL = 64LL * ANCH_PER;      // 3,096,768
  float* out_anc = out;
  float* out_cxp = out + ANCH_TOTAL;
  float* out_cxn = out + ANCH_TOTAL + 64 * 17 * 17;

  // ---- 1) broadcast anchor to output
  bcast_anchor<<<2048, 256, 0, stream>>>(anchor, out_anc, ANCH_PER, ANCH_TOTAL);

  // ---- 2) pack all conv weights into B-fragment order
  for (int l = 0; l < 5; ++l) {
    const LayerCfg& c = LC[l];
    int threads = (c.NPAD / 16) * (c.KPAD / 32) * 32;
    pack_weights<<<(threads + 255) / 256, 256, 0, stream>>>(
        (const float*)d_in[3 + l], P.wfrag + WOFF[l], c.CIN, c.COUT, c.NPAD, c.KPAD);
  }

  // ---- 3) anchor branch (B=1: identical copies => identical BN stats/features)
  run_branch(stream, anchor, 1, 127, 127, P, d_in);
  // final features: convOut[144, 128] = [12,12,128] NHWC -> stash
  hipMemcpyAsync(P.ancF, P.convOut, (size_t)12 * 12 * 128 * sizeof(float),
                 hipMemcpyDeviceToDevice, stream);

  // ---- 4) pos branch + cross-correlation
  run_branch(stream, pos, 64, 255, 255, P, d_in);
  xcorr_kernel<<<64 * 289, 256, 0, stream>>>(P.convOut, P.ancF, out_cxp);

  // ---- 5) neg branch + cross-correlation
  run_branch(stream, neg, 64, 255, 255, P, d_in);
  xcorr_kernel<<<64 * 289, 256, 0, stream>>>(P.convOut, P.ancF, out_cxn);
}
